// SpatialTransformer_9251359556094
// MI455X (gfx1250) — matrix operand order repaired and checked by reading, not compiled, not run on hardware
//
#include <hip/hip_runtime.h>
#include <cstdint>

// ---------------------------------------------------------------------------
// SpatialTransformer block for MI455X (gfx1250, wave32, WMMA 16x16x32 f16)
// ---------------------------------------------------------------------------

typedef _Float16 f16;
typedef __attribute__((ext_vector_type(8)))  _Float16 f16x8;
typedef __attribute__((ext_vector_type(16))) _Float16 f16x16;
typedef __attribute__((ext_vector_type(8)))  float    f32x8;

#define NTOK   2304          // H*W
#define NTOT   4608          // B * NTOK
#define CIN    512
#define NHEAD  8
#define DHEAD  64
#define CTXN   77
#define CTXNP  96            // padded (multiple of 32)
#define CTXD   768
#define FFI    2048
#define ATT_SCALE 0.125f     // 64^-0.5

__device__ __forceinline__ f32x8 wmma16(f16x16 a, f16x16 b, f32x8 c) {
  return __builtin_amdgcn_wmma_f32_16x16x32_f16(false, a, false, b, (short)0, c,
                                                false, false);
}

// B-operand fragment: 16 contiguous halves starting at p (N = lane%16 row of a
// transposed (N,K) matrix; K = kb + (lane<16?0:16) + e folded into p by caller)
__device__ __forceinline__ f16x16 frag_b(const f16* p) {
  f16x8 lo = *(const f16x8*)p;
  f16x8 hi = *(const f16x8*)(p + 8);
  f16x16 r;
#pragma unroll
  for (int i = 0; i < 8; ++i) { r[i] = lo[i]; r[8 + i] = hi[i]; }
  return r;
}

// A-operand fragment from a row-major row pointer.
// lanes 0-15: e0-15 -> K=kb+e
// lanes 16-31: e0-15 -> K=kb+16+e
__device__ __forceinline__ f16x16 frag_a(const f16* rowp, int kb, int hi) {
  f16x8 lo = *(const f16x8*)(rowp + kb + hi * 16);
  f16x8 hv = *(const f16x8*)(rowp + kb + hi * 16 + 8);
  f16x16 r;
#pragma unroll
  for (int i = 0; i < 8; ++i) { r[i] = lo[i]; r[8 + i] = hv[i]; }
  return r;
}

__device__ __forceinline__ void zero4(f32x8 a[4]) {
#pragma unroll
  for (int t = 0; t < 4; ++t)
#pragma unroll
    for (int r = 0; r < 8; ++r) a[t][r] = 0.0f;
}

// Wave-level 16x64 accumulate: A row-major (ldA), Wt row-major (M x K).
__device__ __forceinline__ void gemm_acc4(const f16* A, const f16* Wt, int ldA,
                                          int K, int row0, int aRowMax, int col0,
                                          f32x8 acc[4]) {
  const int lane = threadIdx.x & 31;
  const int l16 = lane & 15, hi = lane >> 4;
  int ar = row0 + l16; if (ar > aRowMax) ar = aRowMax;
  const f16* ap = A + (size_t)ar * ldA;
  const f16* wp = Wt + (size_t)(col0 + l16) * K + hi * 16;
  for (int kb = 0; kb < K; kb += 32) {
    f16x16 af = frag_a(ap, kb, hi);
#pragma unroll
    for (int t = 0; t < 4; ++t) {
      f16x16 bf = frag_b(wp + (size_t)(t * 16) * K + kb);
      acc[t] = wmma16(af, bf, acc[t]);
    }
  }
}

// ------------------------------ small kernels ------------------------------

__global__ void k_wt(const float* __restrict__ w, f16* __restrict__ wt, int K, int M) {
  int idx = blockIdx.x * 256 + threadIdx.x;
  if (idx >= K * M) return;
  int k = idx / M, m = idx % M;
  wt[(size_t)m * K + k] = (f16)w[idx];
}

__global__ void k_f32_to_f16(const float* __restrict__ a, f16* __restrict__ o, int n) {
  int i = blockIdx.x * 256 + threadIdx.x;
  if (i < n) o[i] = (f16)a[i];
}

__global__ void k_gn_stats(const float* __restrict__ x, float* __restrict__ mean,
                           float* __restrict__ rstd) {
  const int bg = blockIdx.x;                 // b*32 + g
  const float* p = x + (size_t)bg * 16 * NTOK;   // group slab is contiguous
  float a0 = 0.f, a1 = 0.f;
  for (int i = threadIdx.x; i < 16 * NTOK; i += 256) {
    float v = p[i]; a0 += v; a1 += v * v;
  }
  __shared__ float ss[256], sq[256];
  ss[threadIdx.x] = a0; sq[threadIdx.x] = a1; __syncthreads();
  for (int s = 128; s > 0; s >>= 1) {
    if (threadIdx.x < s) { ss[threadIdx.x] += ss[threadIdx.x + s];
                           sq[threadIdx.x] += sq[threadIdx.x + s]; }
    __syncthreads();
  }
  if (threadIdx.x == 0) {
    float m = ss[0] * (1.f / (16.f * NTOK));
    float v = sq[0] * (1.f / (16.f * NTOK)) - m * m;
    mean[bg] = m; rstd[bg] = rsqrtf(v + 1e-6f);
  }
}

__global__ void k_gn_apply(const float* __restrict__ x, const float* __restrict__ mean,
                           const float* __restrict__ rstd, const float* __restrict__ gw,
                           const float* __restrict__ gb, f16* __restrict__ h) {
  int idx = blockIdx.x * 256 + threadIdx.x;      // over B*C*NTOK = 2359296
  int b = idx / (CIN * NTOK);
  int rem = idx - b * (CIN * NTOK);
  int c = rem / NTOK, n = rem - c * NTOK;
  int bg = b * 32 + (c >> 4);
  float v = (x[idx] - mean[bg]) * rstd[bg] * gw[c] + gb[c];
  h[((size_t)b * NTOK + n) * CIN + c] = (f16)v;   // transposed to (b, n, c)
}

__global__ __launch_bounds__(128)
void k_ln(const float* __restrict__ X, const float* __restrict__ g,
          const float* __restrict__ b, f16* __restrict__ out) {
  const int row = blockIdx.x;
  const float* x = X + (size_t)row * CIN;
  float a0 = 0.f, a1 = 0.f;
  for (int i = threadIdx.x; i < CIN; i += 128) { float v = x[i]; a0 += v; a1 += v * v; }
  __shared__ float ss[128], sq[128];
  ss[threadIdx.x] = a0; sq[threadIdx.x] = a1; __syncthreads();
  for (int s = 64; s > 0; s >>= 1) {
    if (threadIdx.x < s) { ss[threadIdx.x] += ss[threadIdx.x + s];
                           sq[threadIdx.x] += sq[threadIdx.x + s]; }
    __syncthreads();
  }
  float mean = ss[0] * (1.f / CIN);
  float var  = sq[0] * (1.f / CIN) - mean * mean;
  float rs   = rsqrtf(var + 1e-5f);
  for (int i = threadIdx.x; i < CIN; i += 128)
    out[(size_t)row * CIN + i] = (f16)((x[i] - mean) * rs * g[i] + b[i]);
}

// ------------------------------ GEMM kernels -------------------------------
// block = 128 threads (4 waves); block tile = 16 rows x 256 cols.

__global__ __launch_bounds__(128)
void k_gemm_h16(const f16* __restrict__ A, const f16* __restrict__ Wt,
                f16* __restrict__ Out, int ldA, int K, int M,
                int validRows, int aRowMax) {
  const int wave = threadIdx.x >> 5;
  const int col0 = blockIdx.y * 256 + wave * 64;
  const int row0 = blockIdx.x * 16;
  f32x8 acc[4]; zero4(acc);
  gemm_acc4(A, Wt, ldA, K, row0, aRowMax, col0, acc);
  const int lane = threadIdx.x & 31, l16 = lane & 15, hi = lane >> 4;
#pragma unroll
  for (int t = 0; t < 4; ++t)
#pragma unroll
    for (int r = 0; r < 8; ++r) {
      int row = row0 + r + hi * 8;
      float v = (row < validRows) ? acc[t][r] : 0.0f;
      Out[(size_t)row * M + col0 + t * 16 + l16] = (f16)v;
    }
}

// f16 transposed store: Out[(b*M + col)*Mpad + j], b = row/jmod, j = row%jmod
__global__ __launch_bounds__(128)
void k_gemm_h16T(const f16* __restrict__ A, const f16* __restrict__ Wt,
                 f16* __restrict__ Out, int ldA, int K, int M,
                 int validRows, int aRowMax, int jmod, int Mpad) {
  const int wave = threadIdx.x >> 5;
  const int col0 = blockIdx.y * 256 + wave * 64;
  const int row0 = blockIdx.x * 16;
  f32x8 acc[4]; zero4(acc);
  gemm_acc4(A, Wt, ldA, K, row0, aRowMax, col0, acc);
  const int lane = threadIdx.x & 31, l16 = lane & 15, hi = lane >> 4;
  const int bb = row0 / jmod;          // jmod is a multiple of 16
  const int j0 = row0 - bb * jmod;
#pragma unroll
  for (int t = 0; t < 4; ++t)
#pragma unroll
    for (int r = 0; r < 8; ++r) {
      int row = row0 + r + hi * 8;
      int j = j0 + r + hi * 8;
      float v = (row < validRows) ? acc[t][r] : 0.0f;
      Out[((size_t)bb * M + col0 + t * 16 + l16) * Mpad + j] = (f16)v;
    }
}

template <bool HAS_RES, bool STORE_H>
__global__ __launch_bounds__(128)
void k_gemm_f32(const f16* __restrict__ A, const f16* __restrict__ Wt,
                const float* __restrict__ bias, const float* __restrict__ res,
                float* __restrict__ outF, f16* __restrict__ outH,
                int ldA, int K, int M) {
  const int wave = threadIdx.x >> 5;
  const int col0 = blockIdx.y * 256 + wave * 64;
  const int row0 = blockIdx.x * 16;
  f32x8 acc[4]; zero4(acc);
  gemm_acc4(A, Wt, ldA, K, row0, NTOT - 1, col0, acc);
  const int lane = threadIdx.x & 31, l16 = lane & 15, hi = lane >> 4;
#pragma unroll
  for (int t = 0; t < 4; ++t) {
    int col = col0 + t * 16 + l16;
    float bi = bias[col];
#pragma unroll
    for (int r = 0; r < 8; ++r) {
      int row = row0 + r + hi * 8;
      size_t idx = (size_t)row * M + col;
      float v = acc[t][r] + bi;
      if (HAS_RES) v += res[idx];
      outF[idx] = v;
      if (STORE_H) outH[idx] = (f16)v;
    }
  }
}

// fused GEGLU: out = (A@Wff1[:, m] + b[m]) * gelu(A@Wff1[:, m+2048] + b[m+2048])
__global__ __launch_bounds__(128)
void k_gemm_geglu(const f16* __restrict__ A, const f16* __restrict__ Wt,
                  const float* __restrict__ bias, f16* __restrict__ Out,
                  int ldA, int K) {
  const int wave = threadIdx.x >> 5;
  const int col0 = blockIdx.y * 256 + wave * 64;
  const int row0 = blockIdx.x * 16;
  const int lane = threadIdx.x & 31, l16 = lane & 15, hi = lane >> 4;
  f32x8 aa[4], gg[4]; zero4(aa); zero4(gg);
  const f16* ap = A + (size_t)(row0 + l16) * ldA;
  const f16* wpA = Wt + (size_t)(col0 + l16) * K + hi * 16;
  const f16* wpG = Wt + (size_t)(col0 + FFI + l16) * K + hi * 16;
  for (int kb = 0; kb < K; kb += 32) {
    f16x16 af = frag_a(ap, kb, hi);
#pragma unroll
    for (int t = 0; t < 4; ++t) {
      f16x16 ba = frag_b(wpA + (size_t)(t * 16) * K + kb);
      aa[t] = wmma16(af, ba, aa[t]);
      f16x16 bg = frag_b(wpG + (size_t)(t * 16) * K + kb);
      gg[t] = wmma16(af, bg, gg[t]);
    }
  }
#pragma unroll
  for (int t = 0; t < 4; ++t) {
    int col = col0 + t * 16 + l16;
    float ba = bias[col], bg = bias[col + FFI];
#pragma unroll
    for (int r = 0; r < 8; ++r) {
      int row = row0 + r + hi * 8;
      float av = aa[t][r] + ba;
      float gv = gg[t][r] + bg;
      float ge = 0.5f * gv * (1.0f + erff(gv * 0.70710678118654752f));
      Out[(size_t)row * FFI + col] = (f16)(av * ge);
    }
  }
}

// final: out[(b*C + col)*NTOK + n] = acc + bias[col] + x_in[...]
__global__ __launch_bounds__(128)
void k_gemm_final(const f16* __restrict__ A, const f16* __restrict__ Wt,
                  const float* __restrict__ bias, const float* __restrict__ xin,
                  float* __restrict__ out, int ldA, int K, int M) {
  const int wave = threadIdx.x >> 5;
  const int col0 = blockIdx.y * 256 + wave * 64;
  const int row0 = blockIdx.x * 16;
  f32x8 acc[4]; zero4(acc);
  gemm_acc4(A, Wt, ldA, K, row0, NTOT - 1, col0, acc);
  const int lane = threadIdx.x & 31, l16 = lane & 15, hi = lane >> 4;
  const int bb = row0 / NTOK;
  const int n0 = row0 - bb * NTOK;
#pragma unroll
  for (int t = 0; t < 4; ++t) {
    int col = col0 + t * 16 + l16;
    float bi = bias[col];
#pragma unroll
    for (int r = 0; r < 8; ++r) {
      int n = n0 + r + hi * 8;
      size_t idx = ((size_t)bb * M + col) * NTOK + n;
      out[idx] = acc[t][r] + bi + xin[idx];
    }
  }
}

// ------------------------- flash attention (per-wave) ----------------------
// One wave handles a 16-row Q tile for one (b, h). K row-major (j, d);
// V pre-transposed: Vt[(b*512 + h*64 + d) * kPad + j].
__global__ __launch_bounds__(128)
void k_attn(const f16* __restrict__ Q, const f16* __restrict__ Kм,
            const f16* __restrict__ Vt, f16* __restrict__ O,
            int kValid, int kPad, int kRowsAlloc) {
  __shared__ __align__(16) f16 Pb[4][16][40];   // stride 40 halves: 16B aligned, conflict-free
  const int lane = threadIdx.x & 31, wave = threadIdx.x >> 5;
  const int l16 = lane & 15, hi = lane >> 4;
  const int b = blockIdx.z, h = blockIdx.y;
  const int qTile = blockIdx.x * 4 + wave;      // 0..143
  const size_t qBase = ((size_t)b * NTOK + qTile * 16) * CIN + h * DHEAD;
  const size_t kBase = (size_t)b * kRowsAlloc * CIN + h * DHEAD;
  const size_t vBase = (size_t)(b * CIN + h * DHEAD) * kPad;

  const f16* qp = Q + qBase + (size_t)l16 * CIN;
  f16x16 aq0 = frag_a(qp, 0, hi);
  f16x16 aq1 = frag_a(qp, 32, hi);

  f32x8 acc[4]; zero4(acc);
  float mrow[8], lrow[8];
#pragma unroll
  for (int r = 0; r < 8; ++r) { mrow[r] = -3.0e38f; lrow[r] = 0.f; }

  for (int j0 = 0; j0 < kPad; j0 += 32) {
    // ---- scores: two 16-wide tiles over 64-deep K dim ----
    f32x8 s0, s1;
    {
      const f16* kp = Kм + kBase + (size_t)(j0 + l16) * CIN;
      f32x8 z; 
#pragma unroll
      for (int r = 0; r < 8; ++r) z[r] = 0.f;
      s0 = wmma16(aq0, frag_b(kp + hi * 16), z);
      s0 = wmma16(aq1, frag_b(kp + 32 + hi * 16), s0);
      const f16* kp2 = Kм + kBase + (size_t)(j0 + 16 + l16) * CIN;
      s1 = wmma16(aq0, frag_b(kp2 + hi * 16), z);
      s1 = wmma16(aq1, frag_b(kp2 + 32 + hi * 16), s1);
    }
    const bool ok0 = (j0 + l16) < kValid;
    const bool ok1 = (j0 + 16 + l16) < kValid;
#pragma unroll
    for (int r = 0; r < 8; ++r) {
      s0[r] = ok0 ? s0[r] * ATT_SCALE : -1.0e30f;
      s1[r] = ok1 ? s1[r] * ATT_SCALE : -1.0e30f;
    }
    // ---- online softmax (row = half-wave; 16 cols per half) ----
#pragma unroll
    for (int r = 0; r < 8; ++r) {
      float mt = fmaxf(s0[r], s1[r]);
      mt = fmaxf(mt, __shfl_xor(mt, 1, 32));
      mt = fmaxf(mt, __shfl_xor(mt, 2, 32));
      mt = fmaxf(mt, __shfl_xor(mt, 4, 32));
      mt = fmaxf(mt, __shfl_xor(mt, 8, 32));
      float mnew = fmaxf(mrow[r], mt);
      float alpha = __expf(mrow[r] - mnew);
      mrow[r] = mnew;
      float p0 = __expf(s0[r] - mnew);
      float p1 = __expf(s1[r] - mnew);
      float ps = p0 + p1;
      ps += __shfl_xor(ps, 1, 32);
      ps += __shfl_xor(ps, 2, 32);
      ps += __shfl_xor(ps, 4, 32);
      ps += __shfl_xor(ps, 8, 32);
      lrow[r] = lrow[r] * alpha + ps;
      acc[0][r] *= alpha; acc[1][r] *= alpha;
      acc[2][r] *= alpha; acc[3][r] *= alpha;
      int prow = r + hi * 8;
      Pb[wave][prow][l16] = (f16)p0;
      Pb[wave][prow][16 + l16] = (f16)p1;
    }
    asm volatile("s_wait_dscnt 0" ::: "memory");   // C-layout -> A-layout via LDS
    f16x16 ap = frag_a(&Pb[wave][l16][0], 0, hi);
    asm volatile("" ::: "memory");
    // ---- O += P @ V (K-dim = 32 keys) ----
#pragma unroll
    for (int t = 0; t < 4; ++t) {
      const f16* vp = Vt + vBase + (size_t)(t * 16 + l16) * kPad + j0 + hi * 16;
      acc[t] = wmma16(ap, frag_b(vp), acc[t]);
    }
  }
  // ---- normalize + store ----
  f16* op = O + qBase;
#pragma unroll
  for (int r = 0; r < 8; ++r) {
    float inv = 1.0f / lrow[r];
    int row = r + hi * 8;
#pragma unroll
    for (int t = 0; t < 4; ++t)
      op[(size_t)row * CIN + t * 16 + l16] = (f16)(acc[t][r] * inv);
  }
}

// ------------------------------- host side ---------------------------------

extern "C" void kernel_launch(void* const* d_in, const int* in_sizes, int n_in,
                              void* d_out, int out_size, void* d_ws, size_t ws_size,
                              hipStream_t stream) {
  (void)in_sizes; (void)n_in; (void)out_size; (void)ws_size;
  const float* x     = (const float*)d_in[0];
  const float* ctx   = (const float*)d_in[1];
  const float* gn_g  = (const float*)d_in[2];
  const float* gn_b  = (const float*)d_in[3];
  const float* w_in  = (const float*)d_in[4];
  const float* b_in  = (const float*)d_in[5];
  const float* ln1g  = (const float*)d_in[6];
  const float* ln1b  = (const float*)d_in[7];
  const float* wq1   = (const float*)d_in[8];
  const float* wk1   = (const float*)d_in[9];
  const float* wv1   = (const float*)d_in[10];
  const float* wo1   = (const float*)d_in[11];
  const float* bo1   = (const float*)d_in[12];
  const float* ln2g  = (const float*)d_in[13];
  const float* ln2b  = (const float*)d_in[14];
  const float* wq2   = (const float*)d_in[15];
  const float* wk2   = (const float*)d_in[16];
  const float* wv2   = (const float*)d_in[17];
  const float* wo2   = (const float*)d_in[18];
  const float* bo2   = (const float*)d_in[19];
  const float* ln3g  = (const float*)d_in[20];
  const float* ln3b  = (const float*)d_in[21];
  const float* wff1  = (const float*)d_in[22];
  const float* bff1  = (const float*)d_in[23];
  const float* wff2  = (const float*)d_in[24];
  const float* bff2  = (const float*)d_in[25];
  const float* w_out = (const float*)d_in[26];
  const float* b_out = (const float*)d_in[27];
  float* out = (float*)d_out;

  uint8_t* wsb = (uint8_t*)d_ws;
  size_t off = 0;
  auto alloc = [&](size_t bytes) -> void* {
    void* p = wsb + off;
    off = (off + bytes + 255) & ~(size_t)255;
    return p;
  };
  f16* wt_in  = (f16*)alloc((size_t)CIN * CIN * 2);
  f16* wt_q1  = (f16*)alloc((size_t)CIN * CIN * 2);
  f16* wt_k1  = (f16*)alloc((size_t)CIN * CIN * 2);
  f16* wt_v1  = (f16*)alloc((size_t)CIN * CIN * 2);
  f16* wt_o1  = (f16*)alloc((size_t)CIN * CIN * 2);
  f16* wt_q2  = (f16*)alloc((size_t)CIN * CIN * 2);
  f16* wt_k2  = (f16*)alloc((size_t)CTXD * CIN * 2);
  f16* wt_v2  = (f16*)alloc((size_t)CTXD * CIN * 2);
  f16* wt_o2  = (f16*)alloc((size_t)CIN * CIN * 2);
  f16* wt_ff1 = (f16*)alloc((size_t)CIN * 4096 * 2);
  f16* wt_ff2 = (f16*)alloc((size_t)FFI * CIN * 2);
  f16* wt_out = (f16*)alloc((size_t)CIN * CIN * 2);
  float* gnm  = (float*)alloc(64 * 4);
  float* gnr  = (float*)alloc(64 * 4);
  f16* h16    = (f16*)alloc((size_t)NTOT * CIN * 2);
  float* t    = (float*)alloc((size_t)NTOT * CIN * 4);
  f16* hn16   = (f16*)alloc((size_t)NTOT * CIN * 2);
  f16* q16    = (f16*)alloc((size_t)NTOT * CIN * 2);
  f16* k16    = (f16*)alloc((size_t)NTOT * CIN * 2);
  f16* vT16   = (f16*)alloc((size_t)2 * CIN * NTOK * 2);
  f16* att16  = (f16*)alloc((size_t)NTOT * CIN * 2);
  f16* ctx16  = (f16*)alloc((size_t)2 * CTXN * CTXD * 2);
  f16* k2_16  = (f16*)alloc((size_t)2 * CTXNP * CIN * 2);
  f16* v2T16  = (f16*)alloc((size_t)2 * CIN * CTXNP * 2);
  f16* g16    = (f16*)alloc((size_t)NTOT * FFI * 2);
  f16* t16    = (f16*)alloc((size_t)NTOT * CIN * 2);

  auto wtx = [&](const float* s, f16* d, int K, int M) {
    k_wt<<<(K * M + 255) / 256, 256, 0, stream>>>(s, d, K, M);
  };
  // weight transpose + f16 convert (one pass; L2-resident afterwards)
  wtx(w_in, wt_in, CIN, CIN);
  wtx(wq1, wt_q1, CIN, CIN);   wtx(wk1, wt_k1, CIN, CIN);
  wtx(wv1, wt_v1, CIN, CIN);   wtx(wo1, wt_o1, CIN, CIN);
  wtx(wq2, wt_q2, CIN, CIN);
  wtx(wk2, wt_k2, CTXD, CIN);  wtx(wv2, wt_v2, CTXD, CIN);
  wtx(wo2, wt_o2, CIN, CIN);
  wtx(wff1, wt_ff1, CIN, 4096);
  wtx(wff2, wt_ff2, FFI, CIN);
  wtx(w_out, wt_out, CIN, CIN);

  // GroupNorm -> h16 (b, n, c) f16
  k_gn_stats<<<64, 256, 0, stream>>>(x, gnm, gnr);
  k_gn_apply<<<(NTOT * CIN) / 256, 256, 0, stream>>>(x, gnm, gnr, gn_g, gn_b, h16);

  const dim3 blk(128);
  const dim3 g512(NTOT / 16, CIN / 256);       // (288, 2)
  // proj_in: t = h16 @ w_in + b_in
  k_gemm_f32<false, false><<<g512, blk, 0, stream>>>(h16, wt_in, b_in, nullptr,
                                                     t, nullptr, CIN, CIN, CIN);
  // --- self attention ---
  k_ln<<<NTOT, blk, 0, stream>>>(t, ln1g, ln1b, hn16);
  k_gemm_h16<<<g512, blk, 0, stream>>>(hn16, wt_q1, q16, CIN, CIN, CIN, NTOT, NTOT - 1);
  k_gemm_h16<<<g512, blk, 0, stream>>>(hn16, wt_k1, k16, CIN, CIN, CIN, NTOT, NTOT - 1);
  k_gemm_h16T<<<g512, blk, 0, stream>>>(hn16, wt_v1, vT16, CIN, CIN, CIN, NTOT,
                                        NTOT - 1, NTOK, NTOK);
  k_attn<<<dim3(36, NHEAD, 2), blk, 0, stream>>>(q16, k16, vT16, att16,
                                                 NTOK, NTOK, NTOK);
  k_gemm_f32<true, false><<<g512, blk, 0, stream>>>(att16, wt_o1, bo1, t,
                                                    t, nullptr, CIN, CIN, CIN);
  // --- cross attention ---
  k_ln<<<NTOT, blk, 0, stream>>>(t, ln2g, ln2b, hn16);
  k_gemm_h16<<<g512, blk, 0, stream>>>(hn16, wt_q2, q16, CIN, CIN, CIN, NTOT, NTOT - 1);
  k_f32_to_f16<<<(2 * CTXN * CTXD + 255) / 256, 256, 0, stream>>>(ctx, ctx16,
                                                                  2 * CTXN * CTXD);
  const dim3 gctx(CTXNP / 16, CIN / 256);      // (6, 2)
  for (int b = 0; b < 2; ++b) {
    const f16* cb = ctx16 + (size_t)b * CTXN * CTXD;
    k_gemm_h16<<<gctx, blk, 0, stream>>>(cb, wt_k2, k2_16 + (size_t)b * CTXNP * CIN,
                                         CTXD, CTXD, CIN, CTXN, CTXN - 1);
    k_gemm_h16T<<<gctx, blk, 0, stream>>>(cb, wt_v2, v2T16 + (size_t)b * CIN * CTXNP,
                                          CTXD, CTXD, CIN, CTXN, CTXN - 1,
                                          CTXNP, CTXNP);
  }
  k_attn<<<dim3(36, NHEAD, 2), blk, 0, stream>>>(q16, k2_16, v2T16, att16,
                                                 CTXN, CTXNP, CTXNP);
  k_gemm_f32<true, false><<<g512, blk, 0, stream>>>(att16, wt_o2, bo2, t,
                                                    t, nullptr, CIN, CIN, CIN);
  // --- GEGLU feed-forward ---
  k_ln<<<NTOT, blk, 0, stream>>>(t, ln3g, ln3b, hn16);
  k_gemm_geglu<<<dim3(NTOT / 16, FFI / 256), blk, 0, stream>>>(hn16, wt_ff1, bff1,
                                                               g16, CIN, CIN);
  k_gemm_f32<true, true><<<g512, blk, 0, stream>>>(g16, wt_ff2, bff2, t,
                                                   t, t16, FFI, FFI, CIN);
  // --- proj_out + input residual, transposed back to (B, C, H, W) ---
  k_gemm_final<<<g512, blk, 0, stream>>>(t16, wt_out, b_out, x, out, CIN, CIN, CIN);
}
